// UniversalCalculator_32469952758378
// MI455X (gfx1250) — matrix operand order repaired and checked by reading, not compiled, not run on hardware
//
#include <hip/hip_runtime.h>
#include <hip/hip_bf16.h>

// MoE SwiGLU MLP, grouped-GEMM formulation for MI455X (gfx1250, wave32, WMMA).
// N=4096 tokens, D=2048, FF=1408, E=8, K=2 -> 8192 routed (token,expert) pairs.
// ~142 GFLOP vs ~0.4 GB min HBM traffic @23.3TB/s => bandwidth-bound; use
// V_WMMA_F32_16X16X32_F16 (f32 accum). TM=32 per block: each B fragment feeds
// two WMMAs, halving weight re-reads from L2 and B-load instruction overhead.

#define N_TOKENS  4096
#define D_MODEL   2048
#define D_FF      1408
#define N_EXPERTS 8
#define TOP_K     2
#define N_ENTRIES (N_TOKENS * TOP_K)   // 8192
#define KTILE     32
#define TM        32                   // token rows per block (2 A-subtiles/wave)
#define TN        128                  // 8 waves x 16 cols per block
#define A_LDA     40                   // padded LDS row (80B): 16B-aligned, no conflicts

typedef __attribute__((ext_vector_type(16))) _Float16     v16h;
typedef __attribute__((ext_vector_type(8)))  float        v8f;
typedef __attribute__((ext_vector_type(4)))  float        v4f;
typedef __attribute__((ext_vector_type(4)))  unsigned int v4u;

union AFrag { v16h h; v4u u4[2]; };

// ---------------- fragment loaders (layouts per CDNA5 ISA §7.12.2) --

// A 16x32 f16 from LDS tile As[TM][A_LDA]; mofs selects the M-subtile (0 / 16).
__device__ inline v16h load_a_frag_lds(const _Float16* As, int lane, int mofs) {
  int m  = (lane & 15) + mofs;
  int kb = (lane >> 4) << 3;              // 0 or 8
  AFrag f;
  f.u4[0] = *(const v4u*)(As + m * A_LDA + kb);        // K = kb..kb+7
  f.u4[1] = *(const v4u*)(As + m * A_LDA + 16 + kb);   // K = 16+kb..16+kb+7
  return f.h;
}

// A 16x32 f16 straight from row-major f16 H scratch (ld = D_FF).
__device__ inline v16h load_a_frag_h(const _Float16* __restrict__ H,
                                     int row_base, int k0, int lane, int mofs) {
  int m  = (lane & 15) + mofs;
  int kb = (lane >> 4) << 3;
  const _Float16* p = H + (size_t)(row_base + m) * D_FF + k0 + kb;
  AFrag f;
  f.u4[0] = *(const v4u*)(p);
  f.u4[1] = *(const v4u*)(p + 16);
  return f.h;
}

// B 32x16 f16 built on the fly from an fp32 row-major [K x N] weight matrix.
// lane%16 = column N, lane/16 selects K octets (0..7,16..23 vs 8..15,24..31). Lanes 0..15 hit
// 64B-contiguous columns per row -> coalesced; cvts merge into v_cvt_pk_f16_f32.
__device__ inline v16h load_b_frag_f32(const float* __restrict__ W,
                                       int k0, int n0, int ld, int lane) {
  int n  = n0 + (lane & 15);
  int kb = k0 + ((lane >> 4) << 3);
  const float* p = W + (size_t)kb * ld + n;
  AFrag f;
#pragma unroll
  for (int i = 0; i < 8; ++i) {
    float w0 = p[(size_t)(2 * i + (i >= 4 ? 8 : 0))     * ld];
    float w1 = p[(size_t)(2 * i + 1 + (i >= 4 ? 8 : 0)) * ld];
    f.h[2 * i]     = (_Float16)w0;
    f.h[2 * i + 1] = (_Float16)w1;
  }
  return f.h;
}

// ---------------- kernel 0: routing lists --------------------------------------

__global__ void moe_route(const int* __restrict__ idx, const float* __restrict__ sc,
                          int* __restrict__ expert_off, int* __restrict__ list_tok,
                          float* __restrict__ list_w) {
  __shared__ int s_cnt[N_EXPERTS];
  __shared__ int s_fill[N_EXPERTS];
  int t = threadIdx.x;
  if (t < N_EXPERTS) s_cnt[t] = 0;
  __syncthreads();
  for (int i = t; i < N_ENTRIES; i += blockDim.x)
    atomicAdd(&s_cnt[idx[i]], 1);
  __syncthreads();
  if (t == 0) {
    int run = 0;
    for (int e = 0; e < N_EXPERTS; ++e) {
      expert_off[e] = run;
      s_fill[e]     = run;
      run += s_cnt[e];
    }
    expert_off[N_EXPERTS] = run;
  }
  __syncthreads();
  for (int i = t; i < N_ENTRIES; i += blockDim.x) {
    int e   = idx[i];
    int pos = atomicAdd(&s_fill[e], 1);
    list_tok[pos] = i >> 1;        // token = entry / TOP_K (TOP_K == 2)
    list_w[pos]   = sc[i];
  }
}

// ---------------- kernel 1: zero output (atomics accumulate into it) -----------

__global__ void zero_out(float* __restrict__ y) {
  size_t i = ((size_t)blockIdx.x * blockDim.x + threadIdx.x) * 4;
  if (i < (size_t)N_TOKENS * D_MODEL) {
    float z4[4] = {0.f, 0.f, 0.f, 0.f};
    *(v4u*)(y + i) = *(const v4u*)z4;  // 16B store of zeros
  }
}

// ---------------- kernel 2: H = silu(X W_gate) * (X W_up), grouped by expert ---

__global__ void __launch_bounds__(256) moe_gemm1(
    const float* __restrict__ x,
    const float* __restrict__ Wg,
    const float* __restrict__ Wu,
    const int*   __restrict__ expert_off,
    const int*   __restrict__ list_tok,
    _Float16*    __restrict__ H) {
  int e   = blockIdx.z;
  int off = expert_off[e];
  int cnt = expert_off[e + 1] - off;
  int m0  = blockIdx.x * TM;
  if (m0 >= cnt) return;                       // dynamic grouped-GEMM early exit
  int n0  = blockIdx.y * TN;

  const float* wg = Wg + (size_t)e * D_MODEL * D_FF;
  const float* wu = Wu + (size_t)e * D_MODEL * D_FF;

  __shared__ _Float16 As[TM * A_LDA];          // 32x32 A tile, padded rows
  __shared__ int s_tok[TM];

  int t = threadIdx.x;
  if (t < TM) {
    int i = m0 + t;
    s_tok[t] = list_tok[off + (i < cnt ? i : cnt - 1)];  // clamp tail rows
  }
  __syncthreads();

  // K-invariant staging pointers: thread t stages 4 consecutive cols of row r.
  int sr = t >> 3;                             // 0..31
  int sc = (t & 7) * 4;                        // 0,4,...,28
  const float* xrow = x + (size_t)s_tok[sr] * D_MODEL + sc;
  _Float16*    asp  = As + sr * A_LDA + sc;

  int lane = t & 31;
  int wave = t >> 5;
  int nw   = n0 + wave * 16;                   // this wave's 16-column subtile

  v8f accg0 = {}, accg1 = {};
  v8f accu0 = {}, accu1 = {};

  for (int k0 = 0; k0 < D_MODEL; k0 += KTILE) {
    __syncthreads();                           // As reuse across iterations
    {
      v4f xv = *(const v4f*)(xrow + k0);       // 16B coalesced x load
      asp[0] = (_Float16)xv.x;
      asp[1] = (_Float16)xv.y;
      asp[2] = (_Float16)xv.z;
      asp[3] = (_Float16)xv.w;
    }
    __syncthreads();

    if (k0 + KTILE < D_MODEL) {                // hint next weight K-tile
      __builtin_prefetch(wg + (size_t)(k0 + KTILE) * D_FF + nw, 0, 1);
      __builtin_prefetch(wu + (size_t)(k0 + KTILE) * D_FF + nw, 0, 1);
    }

    v16h a0 = load_a_frag_lds(As, lane, 0);
    v16h a1 = load_a_frag_lds(As, lane, 16);
    v16h bg = load_b_frag_f32(wg, k0, nw, D_FF, lane);
    v16h bu = load_b_frag_f32(wu, k0, nw, D_FF, lane);
    accg0 = __builtin_amdgcn_wmma_f32_16x16x32_f16(false, a0, false, bg,
                                                   (short)0, accg0, false, false);
    accg1 = __builtin_amdgcn_wmma_f32_16x16x32_f16(false, a1, false, bg,
                                                   (short)0, accg1, false, false);
    accu0 = __builtin_amdgcn_wmma_f32_16x16x32_f16(false, a0, false, bu,
                                                   (short)0, accu0, false, false);
    accu1 = __builtin_amdgcn_wmma_f32_16x16x32_f16(false, a1, false, bu,
                                                   (short)0, accu1, false, false);
  }

  // SwiGLU in-register, store H tile (f16 row-major scratch).
  int n  = lane & 15;
  int mh = (lane >> 4) * 8;                    // C/D layout: M = r + 8*(lane>=16)
#pragma unroll
  for (int sub = 0; sub < 2; ++sub) {
#pragma unroll
    for (int r = 0; r < 8; ++r) {
      int m = sub * 16 + r + mh;
      if (m0 + m < cnt) {
        float g = sub ? accg1[r] : accg0[r];
        float u = sub ? accu1[r] : accu0[r];
        float h = g * u / (1.0f + __expf(-g)); // silu(g) * u
        H[(size_t)(off + m0 + m) * D_FF + nw + n] = (_Float16)h;
      }
    }
  }
}

// ---------------- kernel 3: Y += score * (H W_down), scatter-combine -----------

__global__ void __launch_bounds__(256) moe_gemm2(
    const _Float16* __restrict__ H,
    const float*    __restrict__ Wd,
    const int*      __restrict__ expert_off,
    const int*      __restrict__ list_tok,
    const float*    __restrict__ list_w,
    float*          __restrict__ y) {
  int e   = blockIdx.z;
  int off = expert_off[e];
  int cnt = expert_off[e + 1] - off;
  int m0  = blockIdx.x * TM;
  if (m0 >= cnt) return;
  int n0  = blockIdx.y * TN;

  const float* wd = Wd + (size_t)e * D_FF * D_MODEL;

  __shared__ int   s_tok[TM];
  __shared__ float s_w[TM];
  int t = threadIdx.x;
  if (t < TM) {
    int i  = m0 + t;
    int ic = i < cnt ? i : cnt - 1;
    s_tok[t] = list_tok[off + ic];
    s_w[t]   = list_w[off + ic];
  }
  __syncthreads();

  int lane = t & 31;
  int wave = t >> 5;
  int nw   = n0 + wave * 16;
  int row_base = off + m0;

  v8f acc0 = {}, acc1 = {};
  for (int k0 = 0; k0 < D_FF; k0 += KTILE) {   // 44 iterations
    if (k0 + KTILE < D_FF)
      __builtin_prefetch(wd + (size_t)(k0 + KTILE) * D_MODEL + nw, 0, 1);
    v16h a0 = load_a_frag_h(H, row_base, k0, lane, 0);
    v16h a1 = load_a_frag_h(H, row_base, k0, lane, 16);
    v16h b  = load_b_frag_f32(wd, k0, nw, D_MODEL, lane);
    acc0 = __builtin_amdgcn_wmma_f32_16x16x32_f16(false, a0, false, b,
                                                  (short)0, acc0, false, false);
    acc1 = __builtin_amdgcn_wmma_f32_16x16x32_f16(false, a1, false, b,
                                                  (short)0, acc1, false, false);
  }

  int n  = lane & 15;
  int mh = (lane >> 4) * 8;
#pragma unroll
  for (int sub = 0; sub < 2; ++sub) {
#pragma unroll
    for (int r = 0; r < 8; ++r) {
      int m = sub * 16 + r + mh;
      if (m0 + m < cnt) {
        float v = (sub ? acc1[r] : acc0[r]) * s_w[m];
        atomicAdd(&y[(size_t)s_tok[m] * D_MODEL + nw + n], v);
      }
    }
  }
}

// ---------------- launcher -----------------------------------------------------

extern "C" void kernel_launch(void* const* d_in, const int* in_sizes, int n_in,
                              void* d_out, int out_size, void* d_ws, size_t ws_size,
                              hipStream_t stream) {
  const float* x        = (const float*)d_in[0];
  const int*   topk_idx = (const int*)d_in[1];
  const float* topk_sc  = (const float*)d_in[2];
  const float* Wg       = (const float*)d_in[3];
  const float* Wu       = (const float*)d_in[4];
  const float* Wd       = (const float*)d_in[5];
  float*       y        = (float*)d_out;

  // Workspace layout (256B-aligned): offsets | token list | score list | H f16
  // (H sized (8192+TM) rows x 1408 f16 ~ 23MB incl. tail-tile slack).
  char* ws = (char*)d_ws;
  int*       expert_off = (int*)(ws);                      // 9 ints
  int*       list_tok   = (int*)(ws + 256);                // 8192 ints
  float*     list_w     = (float*)(ws + 256 + 32768);      // 8192 floats
  _Float16*  H          = (_Float16*)(ws + 256 + 65536);

  hipLaunchKernelGGL(moe_route, dim3(1), dim3(256), 0, stream,
                     topk_idx, topk_sc, expert_off, list_tok, list_w);
  hipLaunchKernelGGL(zero_out,
                     dim3((unsigned)(((size_t)N_TOKENS * D_MODEL / 4 + 255) / 256)),
                     dim3(256), 0, stream, y);
  hipLaunchKernelGGL(moe_gemm1,
                     dim3(N_ENTRIES / TM, D_FF / TN, N_EXPERTS),
                     dim3(256), 0, stream, x, Wg, Wu, expert_off, list_tok, H);
  hipLaunchKernelGGL(moe_gemm2,
                     dim3(N_ENTRIES / TM, D_MODEL / TN, N_EXPERTS),
                     dim3(256), 0, stream, H, Wd, expert_off, list_tok, list_w, y);
}
